// PLE_27238682591702
// MI455X (gfx1250) — matrix operand order repaired and checked by reading, not compiled, not run on hardware
//
#include <hip/hip_runtime.h>
#include <hip/hip_bf16.h>

// ---------------------------------------------------------------------------
// PLE / CGC multi-task MoE on gfx1250 (MI455X), f16 WMMA with f32 accumulate.
// B=32768, IN=256, D=64, G=32, S=4, E=3, T=3, NEXP=7
// ---------------------------------------------------------------------------

typedef __attribute__((ext_vector_type(16))) _Float16 v16h;
typedef __attribute__((ext_vector_type(8)))  _Float16 v8h;
typedef __attribute__((ext_vector_type(8)))  float    v8f;

#define PLE_B      32768
#define PLE_T      3
#define PLE_WAVES  4     // waves per block in GEMM kernels (wave32)

// ---------------------------------------------------------------------------
// Fused expert FFN: out = relu( relu(A @ W1 + b1) @ W2 + b2 )
// 13 experts (blockIdx.y): 0..3 shared (input Ash), 4..12 task (input Atask + t*stride)
// W1t: [13][64][K] f16 transposed, W2t: [13][64][64] f16 transposed
// out: [13][B][64] f16
// ---------------------------------------------------------------------------
__global__ __launch_bounds__(PLE_WAVES * 32)
void ple_ffn_fused(const _Float16* __restrict__ Ash,
                   const _Float16* __restrict__ Atask,
                   long taskStride, int K,
                   const _Float16* __restrict__ W1t,
                   const float* __restrict__ b1sh, const float* __restrict__ b1te,
                   const _Float16* __restrict__ W2t,
                   const float* __restrict__ b2sh, const float* __restrict__ b2te,
                   _Float16* __restrict__ out)
{
    __shared__ _Float16 tile[PLE_WAVES][16 * 64];

    const int lane = threadIdx.x & 31;
    const int wave = threadIdx.x >> 5;
    const int e    = blockIdx.y;            // expert 0..12
    const int row0 = (blockIdx.x * PLE_WAVES + wave) * 16;
    const int g    = lane >> 4;             // lane group 0/1
    const int ln   = lane & 15;             // M (for A) / N (for B,C)

    const _Float16* A  = (e < 4) ? Ash : (Atask + (long)((e - 4) / 3) * taskStride);
    const float*    b1 = (e < 4) ? (b1sh + e * 64) : (b1te + (e - 4) * 64);
    const float*    b2 = (e < 4) ? (b2sh + e * 64) : (b2te + (e - 4) * 64);
    const _Float16* W1e = W1t + (long)e * 64 * K;
    const _Float16* W2e = W2t + (long)e * 64 * 64;

    v8f acc[4];
    #pragma unroll
    for (int nt = 0; nt < 4; ++nt)
        #pragma unroll
        for (int i = 0; i < 8; ++i) acc[nt][i] = 0.0f;

    // ---- GEMM 1: [16 x K] @ [K x 64] ----
    const _Float16* Arow = A + (long)(row0 + ln) * K;
    for (int k0 = 0; k0 < K; k0 += 32) {
        v16h a;
        {
            v8h lo = *(const v8h*)(Arow + k0 + 16 * g);       // K = k0 + 16g + 0..7
            v8h hi = *(const v8h*)(Arow + k0 + 16 * g + 8);   // K = k0 + 16g + 8..15
            #pragma unroll
            for (int i = 0; i < 8; ++i) { a[i] = lo[i]; a[8 + i] = hi[i]; }
        }
        #pragma unroll
        for (int nt = 0; nt < 4; ++nt) {
            // lane's B fragment: column N = nt*16+ln, K = k0 + 16g + 0..15 (contiguous)
            v16h b = *(const v16h*)(W1e + (long)(nt * 16 + ln) * K + k0 + 16 * g);
            acc[nt] = __builtin_amdgcn_wmma_f32_16x16x32_f16(
                false, a, false, b, (short)0, acc[nt], false, false);
        }
    }

    // bias + relu, stage 16x64 f16 tile in LDS
    _Float16* T0 = tile[wave];
    #pragma unroll
    for (int nt = 0; nt < 4; ++nt) {
        float bv = b1[nt * 16 + ln];
        #pragma unroll
        for (int v = 0; v < 8; ++v) {
            float x = acc[nt][v] + bv;
            x = x > 0.0f ? x : 0.0f;
            T0[(v + 8 * g) * 64 + nt * 16 + ln] = (_Float16)x;  // row M = v+8g, col N
        }
    }
    __syncthreads();

    // ---- GEMM 2: [16 x 64] @ [64 x 64] from LDS ----
    v8f acc2[4];
    #pragma unroll
    for (int nt = 0; nt < 4; ++nt)
        #pragma unroll
        for (int i = 0; i < 8; ++i) acc2[nt][i] = 0.0f;

    #pragma unroll
    for (int k0 = 0; k0 < 64; k0 += 32) {
        v16h a;
        {
            const _Float16* ap = T0 + ln * 64 + k0 + 16 * g;
            v8h lo = *(const v8h*)ap;
            v8h hi = *(const v8h*)(ap + 8);
            #pragma unroll
            for (int i = 0; i < 8; ++i) { a[i] = lo[i]; a[8 + i] = hi[i]; }
        }
        #pragma unroll
        for (int nt = 0; nt < 4; ++nt) {
            v16h b = *(const v16h*)(W2e + (long)(nt * 16 + ln) * 64 + k0 + 16 * g);
            acc2[nt] = __builtin_amdgcn_wmma_f32_16x16x32_f16(
                false, a, false, b, (short)0, acc2[nt], false, false);
        }
    }
    __syncthreads();

    // bias + relu, restage, cooperative coalesced store (tile rows are contiguous)
    #pragma unroll
    for (int nt = 0; nt < 4; ++nt) {
        float bv = b2[nt * 16 + ln];
        #pragma unroll
        for (int v = 0; v < 8; ++v) {
            float x = acc2[nt][v] + bv;
            x = x > 0.0f ? x : 0.0f;
            T0[(v + 8 * g) * 64 + nt * 16 + ln] = (_Float16)x;
        }
    }
    __syncthreads();

    _Float16* outBase = out + ((long)e * PLE_B + row0) * 64;
    #pragma unroll
    for (int it = 0; it < 4; ++it) {
        int off = it * 256 + lane * 8;
        *(v8h*)(outBase + off) = *(const v8h*)(T0 + off);
    }
}

// ---------------------------------------------------------------------------
// Gate hidden GEMM: g1[t] = relu(A[t] @ gW1[t] + gb1[t]),  K -> 32
// G1t: [3][32][K] f16 transposed; out: [3][B][32] f16
// ---------------------------------------------------------------------------
__global__ __launch_bounds__(PLE_WAVES * 32)
void ple_gate_gemm(const _Float16* __restrict__ Abase, long taskStride, int K,
                   const _Float16* __restrict__ G1t,
                   const float* __restrict__ gb1,
                   _Float16* __restrict__ out)
{
    __shared__ _Float16 tile[PLE_WAVES][16 * 32];

    const int lane = threadIdx.x & 31;
    const int wave = threadIdx.x >> 5;
    const int t    = blockIdx.y;
    const int row0 = (blockIdx.x * PLE_WAVES + wave) * 16;
    const int g    = lane >> 4;
    const int ln   = lane & 15;

    const _Float16* A = Abase + (long)t * taskStride;
    const _Float16* W = G1t + (long)t * 32 * K;

    v8f acc[2];
    #pragma unroll
    for (int nt = 0; nt < 2; ++nt)
        #pragma unroll
        for (int i = 0; i < 8; ++i) acc[nt][i] = 0.0f;

    const _Float16* Arow = A + (long)(row0 + ln) * K;
    for (int k0 = 0; k0 < K; k0 += 32) {
        v16h a;
        {
            v8h lo = *(const v8h*)(Arow + k0 + 16 * g);
            v8h hi = *(const v8h*)(Arow + k0 + 16 * g + 8);
            #pragma unroll
            for (int i = 0; i < 8; ++i) { a[i] = lo[i]; a[8 + i] = hi[i]; }
        }
        #pragma unroll
        for (int nt = 0; nt < 2; ++nt) {
            v16h b = *(const v16h*)(W + (long)(nt * 16 + ln) * K + k0 + 16 * g);
            acc[nt] = __builtin_amdgcn_wmma_f32_16x16x32_f16(
                false, a, false, b, (short)0, acc[nt], false, false);
        }
    }

    _Float16* T0 = tile[wave];
    #pragma unroll
    for (int nt = 0; nt < 2; ++nt) {
        float bv = gb1[t * 32 + nt * 16 + ln];
        #pragma unroll
        for (int v = 0; v < 8; ++v) {
            float x = acc[nt][v] + bv;
            x = x > 0.0f ? x : 0.0f;
            T0[(v + 8 * g) * 32 + nt * 16 + ln] = (_Float16)x;
        }
    }
    __syncthreads();

    _Float16* outBase = out + ((long)t * PLE_B + row0) * 32;
    #pragma unroll
    for (int it = 0; it < 2; ++it) {
        int off = it * 256 + lane * 8;
        *(v8h*)(outBase + off) = *(const v8h*)(T0 + off);
    }
}

// ---------------------------------------------------------------------------
// Gate logits + softmax: gate[t,b,:] = softmax(g1[t,b,:] @ gW2[t] + gb2[t])
// gate stored [3][B][8] f32 (7 used, slot 7 zero) for aligned loads.
// ---------------------------------------------------------------------------
__global__ __launch_bounds__(256)
void ple_gate_softmax(const _Float16* __restrict__ g1,
                      const float* __restrict__ gW2,   // [3][32][7]
                      const float* __restrict__ gb2,   // [3][7]
                      float* __restrict__ gate)        // [3][B][8]
{
    long tid = (long)blockIdx.x * 256 + threadIdx.x;   // over T*B
    if (tid >= (long)PLE_T * PLE_B) return;
    int  t = (int)(tid / PLE_B);
    const _Float16* h = g1 + tid * 32;

    float hv[32];
    #pragma unroll
    for (int i = 0; i < 32; i += 8) {
        v8h v = *(const v8h*)(h + i);
        #pragma unroll
        for (int j = 0; j < 8; ++j) hv[i + j] = (float)v[j];
    }

    float logit[7];
    #pragma unroll
    for (int j = 0; j < 7; ++j) {
        float s = gb2[t * 7 + j];
        for (int k = 0; k < 32; ++k) s += hv[k] * gW2[(t * 32 + k) * 7 + j];
        logit[j] = s;
    }
    float m = logit[0];
    #pragma unroll
    for (int j = 1; j < 7; ++j) m = logit[j] > m ? logit[j] : m;
    float ex[7], sum = 0.0f;
    #pragma unroll
    for (int j = 0; j < 7; ++j) { ex[j] = __expf(logit[j] - m); sum += ex[j]; }
    float inv = 1.0f / sum;

    float* gp = gate + tid * 8;
    #pragma unroll
    for (int j = 0; j < 7; ++j) gp[j] = ex[j] * inv;
    gp[7] = 0.0f;
}

// ---------------------------------------------------------------------------
// Combine: xt[t,b,:] = sum_{e<4} gate[t,b,e]*h2[e,b,:] + sum_{e} gate[t,b,4+e]*h2[4+3t+e,b,:]
// One thread handles one (t,b) x 16-column chunk.
// ---------------------------------------------------------------------------
__global__ __launch_bounds__(256)
void ple_combine(const _Float16* __restrict__ h2,    // [13][B][64]
                 const float* __restrict__ gate,     // [3][B][8]
                 _Float16* __restrict__ xt)          // [3][B][64]
{
    long tid = (long)blockIdx.x * 256 + threadIdx.x; // over T*B*4
    if (tid >= (long)PLE_T * PLE_B * 4) return;
    int  c = (int)(tid & 3);
    long r = tid >> 2;                               // t*B + b
    int  t = (int)(r / PLE_B);
    long b = r % PLE_B;

    const float* gp = gate + r * 8;
    float acc[16];
    #pragma unroll
    for (int i = 0; i < 16; ++i) acc[i] = 0.0f;

    #pragma unroll
    for (int e = 0; e < 7; ++e) {
        int xid = (e < 4) ? e : (4 + t * 3 + (e - 4));
        float w = gp[e];
        const _Float16* hp = h2 + ((long)xid * PLE_B + b) * 64 + c * 16;
        v8h a0 = *(const v8h*)hp;
        v8h a1 = *(const v8h*)(hp + 8);
        #pragma unroll
        for (int i = 0; i < 8; ++i) {
            acc[i]     += w * (float)a0[i];
            acc[8 + i] += w * (float)a1[i];
        }
    }

    _Float16* op = xt + r * 64 + c * 16;
    v8h o0, o1;
    #pragma unroll
    for (int i = 0; i < 8; ++i) { o0[i] = (_Float16)acc[i]; o1[i] = (_Float16)acc[8 + i]; }
    *(v8h*)op = o0;
    *(v8h*)(op + 8) = o1;
}

// xm = mean over tasks of xt1
__global__ __launch_bounds__(256)
void ple_mean(const _Float16* __restrict__ xt1, _Float16* __restrict__ xm)
{
    long i = ((long)blockIdx.x * 256 + threadIdx.x) * 8;
    if (i >= (long)PLE_B * 64) return;
    const long S = (long)PLE_B * 64;
    v8h a = *(const v8h*)(xt1 + i);
    v8h b = *(const v8h*)(xt1 + S + i);
    v8h c = *(const v8h*)(xt1 + 2 * S + i);
    v8h o;
    #pragma unroll
    for (int j = 0; j < 8; ++j)
        o[j] = (_Float16)(((float)a[j] + (float)b[j] + (float)c[j]) * (1.0f / 3.0f));
    *(v8h*)(xm + i) = o;
}

// ---------------------------------------------------------------------------
// Tower: out[t,b] = sigmoid( relu(x @ twW1[t] + twb1[t]) @ twW2[t] + twb2[t] )
// Weight indices are block-uniform -> scalar loads; per-lane work is VALU FMA.
// ---------------------------------------------------------------------------
__global__ __launch_bounds__(256)
void ple_tower(const _Float16* __restrict__ xt2,  // [3][B][64]
               const float* __restrict__ W1,      // [3][64][32]
               const float* __restrict__ b1,      // [3][32]
               const float* __restrict__ W2,      // [3][32][1]
               const float* __restrict__ b2,      // [3][1]
               float* __restrict__ out)           // [3*B]
{
    long b = (long)blockIdx.x * 256 + threadIdx.x;
    int  t = blockIdx.y;
    const _Float16* xp = xt2 + ((long)t * PLE_B + b) * 64;

    float xr[64];
    #pragma unroll
    for (int i = 0; i < 64; i += 8) {
        v8h v = *(const v8h*)(xp + i);
        #pragma unroll
        for (int j = 0; j < 8; ++j) xr[i + j] = (float)v[j];
    }

    float acc = b2[t];
    for (int j = 0; j < 32; ++j) {
        float h = b1[t * 32 + j];
        #pragma unroll 8
        for (int k = 0; k < 64; ++k) h += xr[k] * W1[(t * 64 + k) * 32 + j];
        h = h > 0.0f ? h : 0.0f;
        acc += h * W2[t * 32 + j];
    }
    out[(long)t * PLE_B + b] = 1.0f / (1.0f + __expf(-acc));
}

// ---------------------------------------------------------------------------
// Prep kernels
// ---------------------------------------------------------------------------
__global__ __launch_bounds__(256)
void ple_cvt_f16(const float* __restrict__ src, _Float16* __restrict__ dst, long n)
{
    long i = (long)blockIdx.x * 256 + threadIdx.x;
    if (i < n) dst[i] = (_Float16)src[i];
}

// src: [nmat][K][N] f32 -> dst: [nmat][N][K] f16
__global__ __launch_bounds__(256)
void ple_transpose_w(const float* __restrict__ src, _Float16* __restrict__ dst,
                     int K, int N, long total)
{
    long i = (long)blockIdx.x * 256 + threadIdx.x;
    if (i >= total) return;
    long kn = (long)K * N;
    long m = i / kn;
    long r = i % kn;
    int  k = (int)(r / N);
    int  n = (int)(r % N);
    dst[m * kn + (long)n * K + k] = (_Float16)src[i];
}

// ---------------------------------------------------------------------------
// Host-side launch
// ---------------------------------------------------------------------------
static inline size_t ple_align(size_t x) { return (x + 255) & ~(size_t)255; }

extern "C" void kernel_launch(void* const* d_in, const int* in_sizes, int n_in,
                              void* d_out, int out_size, void* d_ws, size_t ws_size,
                              hipStream_t stream) {
    (void)in_sizes; (void)n_in; (void)out_size; (void)ws_size;

    const float* x        = (const float*)d_in[0];
    const float* l0_shW1  = (const float*)d_in[1];
    const float* l0_shb1  = (const float*)d_in[2];
    const float* l0_shW2  = (const float*)d_in[3];
    const float* l0_shb2  = (const float*)d_in[4];
    const float* l0_teW1  = (const float*)d_in[5];
    const float* l0_teb1  = (const float*)d_in[6];
    const float* l0_teW2  = (const float*)d_in[7];
    const float* l0_teb2  = (const float*)d_in[8];
    const float* l0_gW1   = (const float*)d_in[9];
    const float* l0_gb1   = (const float*)d_in[10];
    const float* l0_gW2   = (const float*)d_in[11];
    const float* l0_gb2   = (const float*)d_in[12];
    const float* l1_shW1  = (const float*)d_in[13];
    const float* l1_shb1  = (const float*)d_in[14];
    const float* l1_shW2  = (const float*)d_in[15];
    const float* l1_shb2  = (const float*)d_in[16];
    const float* l1_teW1  = (const float*)d_in[17];
    const float* l1_teb1  = (const float*)d_in[18];
    const float* l1_teW2  = (const float*)d_in[19];
    const float* l1_teb2  = (const float*)d_in[20];
    const float* l1_gW1   = (const float*)d_in[21];
    const float* l1_gb1   = (const float*)d_in[22];
    const float* l1_gW2   = (const float*)d_in[23];
    const float* l1_gb2   = (const float*)d_in[24];
    const float* twW1     = (const float*)d_in[25];
    const float* twb1     = (const float*)d_in[26];
    const float* twW2     = (const float*)d_in[27];
    const float* twb2     = (const float*)d_in[28];
    float* out = (float*)d_out;

    const long B = PLE_B;
    char* ws = (char*)d_ws;

    // workspace layout (bytes)
    size_t off = 0;
    size_t OFF_XH   = off; off = ple_align(off + (size_t)B * 256 * 2);       // x f16
    size_t OFF_H2   = off; off = ple_align(off + (size_t)13 * B * 64 * 2);   // expert outputs
    size_t OFF_G1   = off; off = ple_align(off + (size_t)3 * B * 32 * 2);    // gate hidden
    size_t OFF_GATE = off; off = ple_align(off + (size_t)3 * B * 8 * 4);     // gate probs
    size_t OFF_XT1  = off; off = ple_align(off + (size_t)3 * B * 64 * 2);
    size_t OFF_XT2  = off; off = ple_align(off + (size_t)3 * B * 64 * 2);
    size_t OFF_XM   = off; off = ple_align(off + (size_t)B * 64 * 2);
    size_t OFF_W1P0 = off; off = ple_align(off + (size_t)13 * 64 * 256 * 2);
    size_t OFF_W2P0 = off; off = ple_align(off + (size_t)13 * 64 * 64 * 2);
    size_t OFF_G1P0 = off; off = ple_align(off + (size_t)3 * 32 * 256 * 2);
    size_t OFF_W1P1 = off; off = ple_align(off + (size_t)13 * 64 * 64 * 2);
    size_t OFF_W2P1 = off; off = ple_align(off + (size_t)13 * 64 * 64 * 2);
    size_t OFF_G1P1 = off; off = ple_align(off + (size_t)3 * 32 * 64 * 2);

    _Float16* xh   = (_Float16*)(ws + OFF_XH);
    _Float16* h2   = (_Float16*)(ws + OFF_H2);
    _Float16* g1   = (_Float16*)(ws + OFF_G1);
    float*    gate = (float*)   (ws + OFF_GATE);
    _Float16* xt1  = (_Float16*)(ws + OFF_XT1);
    _Float16* xt2  = (_Float16*)(ws + OFF_XT2);
    _Float16* xm   = (_Float16*)(ws + OFF_XM);
    _Float16* W1P0 = (_Float16*)(ws + OFF_W1P0);
    _Float16* W2P0 = (_Float16*)(ws + OFF_W2P0);
    _Float16* G1P0 = (_Float16*)(ws + OFF_G1P0);
    _Float16* W1P1 = (_Float16*)(ws + OFF_W1P1);
    _Float16* W2P1 = (_Float16*)(ws + OFF_W2P1);
    _Float16* G1P1 = (_Float16*)(ws + OFF_G1P1);

    // ---- prep: convert x, transpose+convert all GEMM weights ----
    {
        long n = B * 256;
        ple_cvt_f16<<<(n + 255) / 256, 256, 0, stream>>>(x, xh, n);
    }
    auto tr = [&](const float* src, _Float16* dst, int K, int N, long total) {
        ple_transpose_w<<<(total + 255) / 256, 256, 0, stream>>>(src, dst, K, N, total);
    };
    tr(l0_shW1, W1P0,                 256, 64, 4L * 256 * 64);
    tr(l0_teW1, W1P0 + 4L * 64 * 256, 256, 64, 9L * 256 * 64);
    tr(l0_shW2, W2P0,                 64,  64, 4L * 64 * 64);
    tr(l0_teW2, W2P0 + 4L * 64 * 64,  64,  64, 9L * 64 * 64);
    tr(l0_gW1,  G1P0,                 256, 32, 3L * 256 * 32);
    tr(l1_shW1, W1P1,                 64,  64, 4L * 64 * 64);
    tr(l1_teW1, W1P1 + 4L * 64 * 64,  64,  64, 9L * 64 * 64);
    tr(l1_shW2, W2P1,                 64,  64, 4L * 64 * 64);
    tr(l1_teW2, W2P1 + 4L * 64 * 64,  64,  64, 9L * 64 * 64);
    tr(l1_gW1,  G1P1,                 64,  32, 3L * 64 * 32);

    dim3 blk(PLE_WAVES * 32);
    dim3 gemmGrid(B / (16 * PLE_WAVES), 13);
    dim3 gateGrid(B / (16 * PLE_WAVES), 3);

    // ---- layer 0 ----
    ple_ffn_fused<<<gemmGrid, blk, 0, stream>>>(
        xh, xh, 0L, 256, W1P0, l0_shb1, l0_teb1, W2P0, l0_shb2, l0_teb2, h2);
    ple_gate_gemm<<<gateGrid, blk, 0, stream>>>(
        xh, 0L, 256, G1P0, l0_gb1, g1);
    {
        long n = 3L * B;
        ple_gate_softmax<<<(n + 255) / 256, 256, 0, stream>>>(g1, l0_gW2, l0_gb2, gate);
    }
    {
        long n = 3L * B * 4;
        ple_combine<<<(n + 255) / 256, 256, 0, stream>>>(h2, gate, xt1);
    }
    {
        long n = B * 64 / 8;
        ple_mean<<<(n + 255) / 256, 256, 0, stream>>>(xt1, xm);
    }

    // ---- layer 1 ----
    ple_ffn_fused<<<gemmGrid, blk, 0, stream>>>(
        xm, xt1, B * 64L, 64, W1P1, l1_shb1, l1_teb1, W2P1, l1_shb2, l1_teb2, h2);
    ple_gate_gemm<<<gateGrid, blk, 0, stream>>>(
        xt1, B * 64L, 64, G1P1, l1_gb1, g1);
    {
        long n = 3L * B;
        ple_gate_softmax<<<(n + 255) / 256, 256, 0, stream>>>(g1, l1_gW2, l1_gb2, gate);
    }
    {
        long n = 3L * B * 4;
        ple_combine<<<(n + 255) / 256, 256, 0, stream>>>(h2, gate, xt2);
    }

    // ---- towers ----
    {
        dim3 tg((unsigned)(B / 256), 3);
        ple_tower<<<tg, 256, 0, stream>>>(xt2, twW1, twb1, twW2, twb2, out);
    }
}
